// VectorQuantizer_23167053595188
// MI455X (gfx1250) — compile-verified, matrix operand order checked by reading, not run on hardware
//
#include <hip/hip_runtime.h>

// ---------------------------------------------------------------------------
// VQ assignment on gfx1250 via bf16-split WMMA.
//   dist^2 = ||x||^2 - 2 x.c + ||c||^2 ;  argmin == argmax (x.c - 0.5||c||^2)
//   x.c computed as x_hi*c_hi + x_hi*c_lo + x_lo*c_hi (bf16 in, fp32 accum)
// Staging uses GLOBAL_LOAD_ASYNC_TO_LDS_B128 (ASYNCcnt) when available.
// ---------------------------------------------------------------------------

typedef __attribute__((ext_vector_type(16))) __bf16 v16bf;
typedef __attribute__((ext_vector_type(8)))  __bf16 v8bf;
typedef __attribute__((ext_vector_type(8)))  float  v8f;

#define DIMV      256      // vector dimensionality
#define MTILE     128      // queries per block (8 waves x 16)
#define NCHUNK    32       // codebook rows staged in LDS per iteration
#define ROWB      264      // LDS row pitch in bf16 elems (528B: 132 dwords -> bank-spread)

#if defined(__has_builtin)
#if __has_builtin(__builtin_amdgcn_global_load_async_to_lds_b128)
#define VQ_ASYNC_LDS 1
#endif
#endif

#ifdef VQ_ASYNC_LDS
// builtin signature (from hipcc diagnostic): (v4i AS1*, v4i AS3*, imm, imm)
typedef int v4i __attribute__((vector_size(16)));
typedef __attribute__((address_space(1))) v4i g_v4i;   // global int4
typedef __attribute__((address_space(3))) v4i l_v4i;   // LDS int4
#endif

// ---------------------------------------------------------------------------
// Prep: codebook fp32 -> (hi, lo) bf16 split + (-0.5 * ||c||^2) fp32.
// One block (256 threads) per codebook row; deterministic reduction.
// ---------------------------------------------------------------------------
__global__ __launch_bounds__(256) void vq_prep_kernel(
    const float* __restrict__ cb, __bf16* __restrict__ chi,
    __bf16* __restrict__ clo, float* __restrict__ nhc) {
  const int code = blockIdx.x;
  const int t = threadIdx.x;
  const float v = cb[code * DIMV + t];
  const __bf16 h = (__bf16)v;
  const __bf16 l = (__bf16)(v - (float)h);
  chi[code * DIMV + t] = h;
  clo[code * DIMV + t] = l;

  float s = v * v;
#pragma unroll
  for (int m = 16; m >= 1; m >>= 1) s += __shfl_xor(s, m, 32);
  __shared__ float ps[8];
  if ((t & 31) == 0) ps[t >> 5] = s;
  __syncthreads();
  if (t == 0) {
    float tot = 0.f;
#pragma unroll
    for (int i = 0; i < 8; ++i) tot += ps[i];
    nhc[code] = -0.5f * tot;
  }
}

// ---------------------------------------------------------------------------
// Main: each wave owns 16 queries (A tile in registers, hi+lo bf16), block
// streams codebook through LDS in 32-row chunks, 3 WMMAs per 32-wide K chunk.
// Running per-lane argmax, final 16-lane shuffle reduction.
// ---------------------------------------------------------------------------
__global__ __launch_bounds__(256, 1) void vq_main_kernel(
    const float* __restrict__ x, const __bf16* __restrict__ chi,
    const __bf16* __restrict__ clo, const float* __restrict__ nhc,
    int* __restrict__ idx_out, int ncodes) {
  const int lane = threadIdx.x & 31;
  const int wave = threadIdx.x >> 5;
  const int l15  = lane & 15;
  const int hi16 = lane >> 4;              // 0: K 0-7/16-23, 1: K 8-15/24-31
  const int kb   = hi16 * 8;
  const int m_base = blockIdx.x * MTILE + wave * 16;
  const int m = m_base + l15;              // row of A this lane feeds

  // ---- build A tile in WMMA 16-bit A layout (once) --------------------------
  v16bf aH[8], aL[8];
#pragma unroll
  for (int c = 0; c < 8; ++c) {
    const float* px = x + (size_t)m * DIMV + c * 32 + kb;
    float xv[16];
    const float4 f0 = *(const float4*)(px + 0);
    const float4 f1 = *(const float4*)(px + 4);
    const float4 f2 = *(const float4*)(px + 16);
    const float4 f3 = *(const float4*)(px + 20);
    xv[0]=f0.x; xv[1]=f0.y; xv[2]=f0.z; xv[3]=f0.w;
    xv[4]=f1.x; xv[5]=f1.y; xv[6]=f1.z; xv[7]=f1.w;
    xv[8]=f2.x; xv[9]=f2.y; xv[10]=f2.z; xv[11]=f2.w;
    xv[12]=f3.x; xv[13]=f3.y; xv[14]=f3.z; xv[15]=f3.w;
#pragma unroll
    for (int i = 0; i < 16; ++i) {
      const __bf16 h = (__bf16)xv[i];
      aH[c][i] = h;
      aL[c][i] = (__bf16)(xv[i] - (float)h);
    }
  }

  float bestV[8];
  int   bestI[8];
#pragma unroll
  for (int r = 0; r < 8; ++r) { bestV[r] = -3.4e38f; bestI[r] = 0; }

  __shared__ __align__(16) __bf16 sHi[NCHUNK * ROWB];
  __shared__ __align__(16) __bf16 sLo[NCHUNK * ROWB];
  __shared__ float sNC[NCHUNK];

  // staging coordinates for this thread (4 x 16B chunks per matrix per iter)
  for (int nc = 0; nc < ncodes; nc += NCHUNK) {
    __syncthreads();  // previous chunk fully consumed
#pragma unroll
    for (int i = 0; i < 4; ++i) {
      const int cid  = i * 256 + threadIdx.x;     // 0..1023 chunks of 8 elems
      const int row  = cid >> 5;
      const int col8 = (cid & 31) << 3;
      const size_t gsrc = (size_t)(nc + row) * DIMV + col8;
      const int    ldst = row * ROWB + col8;
#ifdef VQ_ASYNC_LDS
      __builtin_amdgcn_global_load_async_to_lds_b128(
          (g_v4i*)&chi[gsrc], (l_v4i*)&sHi[ldst], 0, 0);
      __builtin_amdgcn_global_load_async_to_lds_b128(
          (g_v4i*)&clo[gsrc], (l_v4i*)&sLo[ldst], 0, 0);
#else
      *(uint4*)&sHi[ldst] = *(const uint4*)&chi[gsrc];
      *(uint4*)&sLo[ldst] = *(const uint4*)&clo[gsrc];
#endif
    }
    if (threadIdx.x < NCHUNK) sNC[threadIdx.x] = nhc[nc + threadIdx.x];
    // prefetch next chunk toward this WGP while we compute on the current one
    if (nc + NCHUNK < ncodes) {
      const size_t pf = (size_t)(nc + NCHUNK) * DIMV + threadIdx.x * 8;
      __builtin_prefetch(&chi[pf], 0, 1);
      __builtin_prefetch(&clo[pf], 0, 1);
    }
#ifdef VQ_ASYNC_LDS
    asm volatile("s_wait_asynccnt 0" ::: "memory");
#endif
    __syncthreads();

#pragma unroll
    for (int nt = 0; nt < NCHUNK / 16; ++nt) {
      const int nl = nt * 16 + l15;               // column this lane owns
      const float nh = sNC[nl];
      v8f acc = {nh, nh, nh, nh, nh, nh, nh, nh}; // seeds -0.5*||c||^2
      const __bf16* bh = &sHi[nl * ROWB + kb];
      const __bf16* bl = &sLo[nl * ROWB + kb];
#pragma unroll
      for (int c = 0; c < 8; ++c) {
        const v8bf h0 = *(const v8bf*)(bh + c * 32);
        const v8bf h1 = *(const v8bf*)(bh + c * 32 + 16);
        const v8bf l0 = *(const v8bf*)(bl + c * 32);
        const v8bf l1 = *(const v8bf*)(bl + c * 32 + 16);
        const v16bf Bh = __builtin_shufflevector(h0, h1,
            0,1,2,3,4,5,6,7,8,9,10,11,12,13,14,15);
        const v16bf Bl = __builtin_shufflevector(l0, l1,
            0,1,2,3,4,5,6,7,8,9,10,11,12,13,14,15);
        acc = __builtin_amdgcn_wmma_f32_16x16x32_bf16(
            false, aH[c], false, Bh, (short)0, acc, false, false);
        acc = __builtin_amdgcn_wmma_f32_16x16x32_bf16(
            false, aH[c], false, Bl, (short)0, acc, false, false);
        acc = __builtin_amdgcn_wmma_f32_16x16x32_bf16(
            false, aL[c], false, Bh, (short)0, acc, false, false);
      }
      const int n = nc + nl;
#pragma unroll
      for (int r = 0; r < 8; ++r) {
        if (acc[r] > bestV[r]) { bestV[r] = acc[r]; bestI[r] = n; }
      }
    }
  }

  // ---- cross-lane argmax within each 16-lane half (ties -> lowest index) ----
#pragma unroll
  for (int r = 0; r < 8; ++r) {
    float v = bestV[r];
    int   bi = bestI[r];
#pragma unroll
    for (int msk = 8; msk >= 1; msk >>= 1) {
      const float ov = __shfl_xor(v, msk, 16);
      const int   oi = __shfl_xor(bi, msk, 16);
      if (ov > v || (ov == v && oi < bi)) { v = ov; bi = oi; }
    }
    if (l15 == 0) idx_out[m_base + r + (hi16 << 3)] = bi;
  }
}

// ---------------------------------------------------------------------------
// Gather: quantized[q] = codebook[idx[q]]
// ---------------------------------------------------------------------------
__global__ __launch_bounds__(256) void vq_gather_kernel(
    const float* __restrict__ cb, const int* __restrict__ idx,
    float* __restrict__ qout) {
  const int q = blockIdx.x;
  const int d = threadIdx.x;
  qout[(size_t)q * DIMV + d] = cb[(size_t)idx[q] * DIMV + d];
}

extern "C" void kernel_launch(void* const* d_in, const int* in_sizes, int n_in,
                              void* d_out, int out_size, void* d_ws, size_t ws_size,
                              hipStream_t stream) {
  const float* x  = (const float*)d_in[0];   // [4,4096,256] f32
  const float* cb = (const float*)d_in[1];   // [8192,256]  f32
  const int nq = in_sizes[0] / DIMV;         // 16384
  const int nk = in_sizes[1] / DIMV;         // 8192

  int*   idx_out = (int*)d_out;              // first nq elements: indices
  float* q_out   = (float*)d_out + nq;       // then nq*256 floats: quantized

  // workspace: c_hi bf16 | c_lo bf16 | -0.5*||c||^2 f32   (~8.4 MB)
  __bf16* chi = (__bf16*)d_ws;
  __bf16* clo = chi + (size_t)nk * DIMV;
  float*  nhc = (float*)((char*)d_ws + (size_t)2 * nk * DIMV * sizeof(__bf16));

  vq_prep_kernel<<<nk, 256, 0, stream>>>(cb, chi, clo, nhc);
  vq_main_kernel<<<nq / MTILE, 256, 0, stream>>>(x, chi, clo, nhc, idx_out, nk);
  vq_gather_kernel<<<nq, 256, 0, stream>>>(cb, idx_out, q_out);
}
